// FNetEncoderLayer_82626580841095
// MI455X (gfx1250) — matrix operand order repaired and checked by reading, not compiled, not run on hardware
//
#include <hip/hip_runtime.h>
#include <hip/hip_bf16.h>
#include <math.h>

typedef __bf16 bf16_t;
typedef __attribute__((ext_vector_type(16))) __bf16 v16bf;
typedef __attribute__((ext_vector_type(8)))  float   v8f;

#if defined(__has_builtin)
#if __has_builtin(__builtin_amdgcn_tensor_load_to_lds) && __has_builtin(__builtin_amdgcn_s_wait_tensorcnt)
#define USE_TDM 1
#endif
#endif

namespace {
constexpr int  Dd = 2524;
constexpr int  Ss = 512;
constexpr int  Bb = 32;
constexpr int  Hh = 4 * Dd;            // 10096
constexpr long Mr = (long)Ss * Bb;     // 16384 rows
constexpr float TWO_PI = 6.28318530717958647692f;
}

#ifdef USE_TDM
typedef unsigned int u32x4 __attribute__((ext_vector_type(4)));
typedef int          i32x4 __attribute__((ext_vector_type(4)));
typedef int          i32x8 __attribute__((ext_vector_type(8)));

// Issue one TDM 2D tile load: tile0 x tile1 elements (2B each) into LDS.
// tensor_dim0/1 = REMAINING extent from the tile origin (HW zero-fills OOB).
// stride0 = elements between consecutive tile rows in memory.
__device__ __forceinline__ void tdm_load_2d(unsigned lds_addr, const void* gptr,
                                            long rem0, long rem1, long stride0,
                                            int tile0, int tile1)
{
    unsigned long long ga = (unsigned long long)gptr;
    unsigned td0 = (unsigned)(rem0 < 0 ? 0 : rem0);
    unsigned td1 = (unsigned)(rem1 < 0 ? 0 : rem1);
    unsigned long long st0 = (unsigned long long)stride0;
    u32x4 g0;
    g0[0] = 1u;                                              // count=1, load-descr
    g0[1] = lds_addr;                                        // LDS byte address
    g0[2] = (unsigned)(ga & 0xffffffffu);                    // global_addr[31:0]
    g0[3] = (unsigned)((ga >> 32) & 0x1ffffffu) | (2u << 30);// addr[56:32] | type=2
    i32x8 g1;
    g1[0] = (int)(1u << 16);                                 // data_size=1 (2 bytes)
    g1[1] = (int)((td0 & 0xffffu) << 16);                    // tensor_dim0[15:0]
    g1[2] = (int)(((td0 >> 16) & 0xffffu) | ((td1 & 0xffffu) << 16));
    g1[3] = (int)(((td1 >> 16) & 0xffffu) | ((unsigned)tile0 << 16));
    g1[4] = (int)(unsigned)tile1;                            // tile_dim1 (tile_dim2=0)
    g1[5] = (int)(unsigned)(st0 & 0xffffffffu);              // dim0_stride[31:0]
    g1[6] = (int)(unsigned)((st0 >> 32) & 0xffffu);          // dim0_stride[47:32]
    g1[7] = 0;
    i32x4 z4 = {0, 0, 0, 0};
#if __clang_major__ >= 23
    i32x8 z8 = {0, 0, 0, 0, 0, 0, 0, 0};
    __builtin_amdgcn_tensor_load_to_lds(g0, g1, z4, z4, z8, 0);
#else
    __builtin_amdgcn_tensor_load_to_lds(g0, g1, z4, z4, 0);
#endif
}
#endif // USE_TDM

// ---------------------------------------------------------------------------
// bf16 WMMA GEMM: C[z] = epi(A[z](MxK) * B[z](KxN) + bias)
//   A: elem(r,k) at A + z*batA + r*ldA + k            (k contiguous)
//   B: elem(k,n) at B + z*batB + n*sBn + k            (k contiguous!)
//   C: elem(r,n) at C + z*batC + r*ldC + n
// EPI: 0 = fp32 out, 1 = bf16 out, 2 = bf16 GELU(exact), 3 = bf16 ReLU
// 256 thr = 8 waves; block tile 128x128x64; wave: 32x64 = 2x4 WMMA frags,
// two k=32 sub-steps per staged tile. Tiles staged by TDM (double buffered)
// or 8B-chunked loads (fallback).
// NOTE: all K used are multiples of 4 and all strides multiples of 4 elems,
// so 8-byte chunks never straddle the K edge and stay 8B-aligned.
// ---------------------------------------------------------------------------
template <int EPI>
__global__ void __launch_bounds__(256)
gemm_bf16_wmma(const bf16_t* __restrict__ A, long batA, long ldA,
               const bf16_t* __restrict__ Bm, long batB, long sBn,
               float* __restrict__ Cf, bf16_t* __restrict__ Cb, long batC, long ldC,
               const float* __restrict__ bias,
               int M, int N, int K)
{
    __shared__ bf16_t As[2][128][64];   // As[buf][m][k]
    __shared__ bf16_t Bs[2][128][64];   // Bs[buf][n][k]  (k contiguous)

    const int tid  = threadIdx.x;
    const int wave = tid >> 5;
    const int lane = tid & 31;
    const int hlf  = lane >> 4;
    const int lidx = lane & 15;
    const int wm   = wave & 3;          // 4 waves along M
    const int wn   = wave >> 2;         // 2 waves along N
    const int m0   = blockIdx.y * 128;
    const int n0   = blockIdx.x * 128;
    const int z    = blockIdx.z;

    A  += (long)z * batA;
    Bm += (long)z * batB;

    v8f acc[2][4] = {};
    const int nk = (K + 63) >> 6;

#ifdef USE_TDM
    if (wave == 0) {
        tdm_load_2d((unsigned)(uintptr_t)&As[0][0][0], A + (long)m0 * ldA,
                    K, M - m0, ldA, 64, 128);
        tdm_load_2d((unsigned)(uintptr_t)&Bs[0][0][0], Bm + (long)n0 * sBn,
                    K, N - n0, sBn, 64, 128);
    }
#endif

    for (int it = 0; it < nk; ++it) {
        const int cur = it & 1;
        const int k0  = it << 6;
#ifdef USE_TDM
        if (wave == 0) {
            if (it + 1 < nk) {   // prefetch next tiles into the other buffer
                const int kn = (it + 1) << 6;
                tdm_load_2d((unsigned)(uintptr_t)&As[cur ^ 1][0][0],
                            A + (long)m0 * ldA + kn, K - kn, M - m0, ldA, 64, 128);
                tdm_load_2d((unsigned)(uintptr_t)&Bs[cur ^ 1][0][0],
                            Bm + (long)n0 * sBn + kn, K - kn, N - n0, sBn, 64, 128);
                __builtin_amdgcn_s_wait_tensorcnt(2); // current pair complete
            } else {
                __builtin_amdgcn_s_wait_tensorcnt(0);
            }
        }
        __syncthreads();
#else
        __syncthreads();
        // 8-byte chunked staging: 2048 chunks per tile, 8 per thread.
        #pragma unroll
        for (int i = 0; i < 8; ++i) {
            int c = tid + i * 256;
            int r = c >> 4, kc = (c & 15) * 4;
            int gm = m0 + r, gk = k0 + kc;
            unsigned long long v = 0ull;
            if (gm < M && gk + 4 <= K)
                v = *(const unsigned long long*)(A + (long)gm * ldA + gk);
            *(unsigned long long*)&As[cur][r][kc] = v;
        }
        #pragma unroll
        for (int i = 0; i < 8; ++i) {
            int c = tid + i * 256;
            int r = c >> 4, kc = (c & 15) * 4;
            int gn = n0 + r, gk = k0 + kc;
            unsigned long long v = 0ull;
            if (gn < N && gk + 4 <= K)
                v = *(const unsigned long long*)(Bm + (long)gn * sBn + gk);
            *(unsigned long long*)&Bs[cur][r][kc] = v;
        }
        __syncthreads();
#endif
        // ---- two k=32 sub-steps; A and B fragments both in contiguous k order (k = 16*half + e) ----
        #pragma unroll
        for (int sub = 0; sub < 2; ++sub) {
            const int kb = sub * 32;
            v16bf afr[2];
            #pragma unroll
            for (int fi = 0; fi < 2; ++fi) {
                const bf16_t* row = &As[cur][wm * 32 + fi * 16 + lidx][kb];
                #pragma unroll
                for (int e = 0; e < 8; ++e) {
                    afr[fi][e]     = row[hlf * 16 + e];       // K = hlf*16 + e
                    afr[fi][8 + e] = row[hlf * 16 + 8 + e];   // K = hlf*16 + 8 + e
                }
            }
            v16bf bfr[4];
            #pragma unroll
            for (int fj = 0; fj < 4; ++fj) {
                const bf16_t* col = &Bs[cur][wn * 64 + fj * 16 + lidx][kb];
                #pragma unroll
                for (int e = 0; e < 16; ++e)
                    bfr[fj][e] = col[hlf * 16 + e];           // K = hlf*16 + e
            }
            #pragma unroll
            for (int fi = 0; fi < 2; ++fi)
                #pragma unroll
                for (int fj = 0; fj < 4; ++fj)
                    acc[fi][fj] = __builtin_amdgcn_wmma_f32_16x16x32_bf16(
                        false, afr[fi], false, bfr[fj], (short)0, acc[fi][fj], false, false);
        }
#ifdef USE_TDM
        __syncthreads();   // tile reads done before next prefetch overwrites
#endif
    }

    if (Cf) Cf += (long)z * batC;
    if (Cb) Cb += (long)z * batC;

    #pragma unroll
    for (int fi = 0; fi < 2; ++fi)
        #pragma unroll
        for (int fj = 0; fj < 4; ++fj) {
            int col = n0 + wn * 64 + fj * 16 + lidx;
            if (col >= N) continue;
            float bv = bias ? bias[col] : 0.0f;
            #pragma unroll
            for (int v = 0; v < 8; ++v) {
                int row = m0 + wm * 32 + fi * 16 + hlf * 8 + v;  // C: M = v + 8*half
                if (row >= M) continue;
                float val = acc[fi][fj][v] + bv;
                if (EPI == 2) val = 0.5f * val * (1.0f + erff(val * 0.70710678118f));
                if (EPI == 3) val = fmaxf(val, 0.0f);
                if (EPI == 0) Cf[(long)row * ldC + col] = val;
                else          Cb[(long)row * ldC + col] = (bf16_t)val;
            }
        }
}

// ---------------------------------------------------------------------------
// Support kernels
// ---------------------------------------------------------------------------
__global__ void __launch_bounds__(256)
cvt_bf16(bf16_t* dst, const float* src, long n) {
    for (long i = (long)blockIdx.x * 256 + threadIdx.x; i < n; i += (long)gridDim.x * 256)
        dst[i] = (bf16_t)src[i];
}

// CD[d'*n + d] = cos(2*pi*d*d'/n), SD = sin(...)  (both symmetric)
__global__ void __launch_bounds__(256)
build_dft(bf16_t* Cd, bf16_t* Sd, int n) {
    long tot = (long)n * n;
    for (long i = (long)blockIdx.x * 256 + threadIdx.x; i < tot; i += (long)gridDim.x * 256) {
        long dp = i / n, d = i - dp * n;
        long m = (dp * d) % n;
        float ang = (float)(6.283185307179586 * (double)m / (double)n);
        Cd[i] = (bf16_t)cosf(ang);
        Sd[i] = (bf16_t)sinf(ang);
    }
}

// ebT[b][d][s] = eb[s][b][d]  (bf16 transpose for the att-einsum B operand)
__global__ void __launch_bounds__(256)
make_ebT(bf16_t* __restrict__ dst, const bf16_t* __restrict__ src) {
    long tot = (long)Bb * Dd * Ss;
    for (long i = (long)blockIdx.x * 256 + threadIdx.x; i < tot; i += (long)gridDim.x * 256) {
        long b = i / ((long)Dd * Ss);
        long rem = i - b * (long)Dd * Ss;
        long d = rem / Ss, s = rem - d * Ss;
        dst[i] = src[(s * Bb + b) * (long)Dd + d];
    }
}

// T[s,b,d] = x[s,b,d] + sum_b' cos(2pi b b'/32)*Y1[s,b',d] - sin(...)*Y2[s,b',d]
// 32x32 cos/sin coefficient matrices precomputed once per block in LDS.
__global__ void __launch_bounds__(256)
mix_residual(float* __restrict__ Tout, const float* __restrict__ Y1,
             const float* __restrict__ Y2, const float* __restrict__ xin, int D) {
    __shared__ float cmat[32][32], smat[32][32];
    for (int i = threadIdx.x; i < 1024; i += 256) {
        int b = i >> 5, bp = i & 31;
        float ang = (TWO_PI / 32.0f) * (float)((b * bp) & 31);
        cmat[b][bp] = __cosf(ang);
        smat[b][bp] = __sinf(ang);
    }
    __syncthreads();
    int d = blockIdx.x * 256 + threadIdx.x;
    int s = blockIdx.y;
    if (d >= D) return;
    long base = (long)s * 32 * D + d;
    float y1[32], y2[32];
    #pragma unroll
    for (int bp = 0; bp < 32; ++bp) {
        y1[bp] = Y1[base + (long)bp * D];
        y2[bp] = Y2[base + (long)bp * D];
    }
    for (int b = 0; b < 32; ++b) {
        float acc = xin[base + (long)b * D];
        #pragma unroll 8
        for (int bp = 0; bp < 32; ++bp) {
            acc = fmaf(cmat[b][bp], y1[bp], acc);
            acc = fmaf(-smat[b][bp], y2[bp], acc);
        }
        Tout[base + (long)b * D] = acc;
    }
}

// RMSNorm row kernel: v = in + res(optional); out = scale*v/(rms+eps)
__global__ void __launch_bounds__(256)
rmsnorm_kernel(const float* __restrict__ in, const float* __restrict__ res,
               const float* __restrict__ scale, float* outf, bf16_t* outb, int n) {
    long row = blockIdx.x;
    const float* p = in + row * (long)n;
    const float* r = res ? res + row * (long)n : nullptr;
    int tid = threadIdx.x;
    float ss = 0.0f;
    for (int i = tid; i < n; i += 256) {
        float v = p[i] + (r ? r[i] : 0.0f);
        ss += v * v;
    }
    __shared__ float red[256];
    red[tid] = ss; __syncthreads();
    for (int s = 128; s > 0; s >>= 1) { if (tid < s) red[tid] += red[tid + s]; __syncthreads(); }
    float rms = sqrtf(red[0]) * rsqrtf((float)n);
    float inv = 1.0f / (rms + 1e-8f);
    for (int i = tid; i < n; i += 256) {
        float v = p[i] + (r ? r[i] : 0.0f);
        float o = scale[i] * v * inv;
        if (outf) outf[row * (long)n + i] = o;
        outb[row * (long)n + i] = (bf16_t)o;
    }
}

// per (t,b): a = softmax(tanh(scores*mask^2)); a *= mask; a /= sum(a)
__global__ void __launch_bounds__(256)
softmax_tanh_mask(const float* __restrict__ scores, const float* __restrict__ mask,
                  float* __restrict__ alpha, bf16_t* __restrict__ ab,
                  int Bn, int Tn, int Sn) {
    int t = blockIdx.x, b = blockIdx.y, tid = threadIdx.x;
    const float* row = scores + ((long)b * Tn + t) * Sn;
    const float* mk  = mask + (long)b * Sn;
    float v[2], mv[2];
    float mx = -1e30f;
    #pragma unroll
    for (int j = 0; j < 2; ++j) {
        int i = tid + j * 256;
        float m = (i < Sn) ? mk[i] : 0.0f;
        float s = (i < Sn) ? row[i] : 0.0f;
        float tv = tanhf(s * m * m);
        v[j] = tv; mv[j] = m;
        if (i < Sn) mx = fmaxf(mx, tv);
    }
    __shared__ float red[256];
    red[tid] = mx; __syncthreads();
    for (int s = 128; s > 0; s >>= 1) { if (tid < s) red[tid] = fmaxf(red[tid], red[tid + s]); __syncthreads(); }
    mx = red[0]; __syncthreads();
    float sum = 0.0f;
    #pragma unroll
    for (int j = 0; j < 2; ++j) {
        int i = tid + j * 256;
        float e = __expf(v[j] - mx);
        v[j] = e;
        if (i < Sn) sum += e;
    }
    red[tid] = sum; __syncthreads();
    for (int s = 128; s > 0; s >>= 1) { if (tid < s) red[tid] += red[tid + s]; __syncthreads(); }
    float s1 = red[0]; __syncthreads();
    float sum2 = 0.0f;
    #pragma unroll
    for (int j = 0; j < 2; ++j) {
        int i = tid + j * 256;
        float a = (v[j] / s1) * mv[j];
        v[j] = a;
        if (i < Sn) sum2 += a;
    }
    red[tid] = sum2; __syncthreads();
    for (int s = 128; s > 0; s >>= 1) { if (tid < s) red[tid] += red[tid + s]; __syncthreads(); }
    float s2 = red[0];
    #pragma unroll
    for (int j = 0; j < 2; ++j) {
        int i = tid + j * 256;
        if (i < Sn) {
            float a = v[j] / s2;
            long o = (long)t * Bn * Sn + (long)b * Sn + i;
            alpha[o] = a;
            ab[o] = (bf16_t)a;
        }
    }
}

// log_prob[row, 0..5] = hid[row,:] @ fc_w.T + fc_b
__global__ void __launch_bounds__(256)
fc_kernel(const bf16_t* __restrict__ hid, const float* __restrict__ fw,
          const float* __restrict__ fb, float* __restrict__ out, int n) {
    long row = blockIdx.x;
    const bf16_t* h = hid + row * (long)n;
    int tid = threadIdx.x;
    float acc[6] = {0, 0, 0, 0, 0, 0};
    for (int i = tid; i < n; i += 256) {
        float hv = (float)h[i];
        #pragma unroll
        for (int c = 0; c < 6; ++c) acc[c] += hv * fw[(long)c * n + i];
    }
    __shared__ float red[6][256];
    #pragma unroll
    for (int c = 0; c < 6; ++c) red[c][tid] = acc[c];
    __syncthreads();
    for (int s = 128; s > 0; s >>= 1) {
        if (tid < s)
            #pragma unroll
            for (int c = 0; c < 6; ++c) red[c][tid] += red[c][tid + s];
        __syncthreads();
    }
    if (tid < 6) out[row * 6 + tid] = red[tid][0] + fb[tid];
}

// ---------------------------------------------------------------------------
extern "C" void kernel_launch(void* const* d_in, const int* in_sizes, int n_in,
                              void* d_out, int out_size, void* d_ws, size_t ws_size,
                              hipStream_t stream) {
    (void)in_sizes; (void)n_in; (void)out_size; (void)ws_size;
    const float* x     = (const float*)d_in[0];
    // d_in[1] = y (unused by reference)
    const float* mask  = (const float*)d_in[2];
    const float* n1s   = (const float*)d_in[3];
    const float* n2s   = (const float*)d_in[4];
    const float* w1    = (const float*)d_in[5];
    const float* b1    = (const float*)d_in[6];
    const float* w2    = (const float*)d_in[7];
    const float* b2    = (const float*)d_in[8];
    const float* att_w = (const float*)d_in[9];
    const float* att_b = (const float*)d_in[10];
    const float* lin_w = (const float*)d_in[11];
    const float* lin_b = (const float*)d_in[12];
    const float* fc_w  = (const float*)d_in[13];
    const float* fc_b  = (const float*)d_in[14];

    char* ws = (char*)d_ws;
    size_t off = 0;
    auto alloc = [&](size_t bytes) -> char* {
        char* p = ws + off;
        off += (bytes + 255) & ~(size_t)255;
        return p;
    };
    bf16_t* xb   = (bf16_t*)alloc(Mr * Dd * 2);
    bf16_t* CDm  = (bf16_t*)alloc((size_t)Dd * Dd * 2);
    bf16_t* SDm  = (bf16_t*)alloc((size_t)Dd * Dd * 2);
    bf16_t* w1b  = (bf16_t*)alloc((size_t)Hh * Dd * 2);   // w1 (H,D) bf16
    bf16_t* w2b  = (bf16_t*)alloc((size_t)Dd * Hh * 2);   // w2 (D,H) bf16
    bf16_t* awb  = (bf16_t*)alloc((size_t)Dd * Dd * 2);   // att_w bf16
    bf16_t* lwb  = (bf16_t*)alloc((size_t)Dd * Dd * 2);   // lin_w bf16
    float*  Y1   = (float*)alloc(Mr * Dd * 4);
    float*  Y2   = (float*)alloc(Mr * Dd * 4);
    float*  Tb   = (float*)alloc(Mr * Dd * 4);
    bf16_t* x1b  = (bf16_t*)alloc(Mr * Dd * 2);
    bf16_t* Hb   = (bf16_t*)alloc(Mr * Hh * 2);
    bf16_t* eb   = (bf16_t*)alloc(Mr * Dd * 2);
    bf16_t* ebT  = (bf16_t*)alloc((size_t)Bb * Dd * Ss * 2);
    bf16_t* qb   = (bf16_t*)alloc(Mr * Dd * 2);
    float*  scor = (float*)alloc((size_t)Bb * Ss * Ss * 4);
    bf16_t* ab   = (bf16_t*)alloc((size_t)Ss * Bb * Ss * 2);
    bf16_t* attb = (bf16_t*)alloc(Mr * Dd * 2);
    bf16_t* hidb = (bf16_t*)alloc(Mr * Dd * 2);
    float*  X1 = Y1;   // alias: Y1 dead after mix
    float*  X2 = Y2;   // alias: Y2 dead after mix

    float* log_prob = (float*)d_out;
    float* alpha    = log_prob + (size_t)Ss * Bb * 6;

    auto cdiv = [](long a, long b) { return (unsigned)((a + b - 1) / b); };
    dim3 blk(256);

    // --- prep: bf16 casts + DFT matrices ---
    cvt_bf16<<<4096, blk, 0, stream>>>(xb, x, Mr * Dd);
    build_dft<<<4096, blk, 0, stream>>>(CDm, SDm, Dd);
    cvt_bf16<<<4096, blk, 0, stream>>>(w1b, w1, (long)Hh * Dd);
    cvt_bf16<<<4096, blk, 0, stream>>>(w2b, w2, (long)Dd * Hh);
    cvt_bf16<<<2048, blk, 0, stream>>>(awb, att_w, (long)Dd * Dd);
    cvt_bf16<<<2048, blk, 0, stream>>>(lwb, lin_w, (long)Dd * Dd);

    // --- FNet: Y1 = x@CD, Y2 = x@SD (DFT matrices are symmetric) ---
    gemm_bf16_wmma<0><<<dim3(cdiv(Dd, 128), cdiv(Mr, 128), 1), blk, 0, stream>>>(
        xb, 0, Dd, CDm, 0, Dd, Y1, nullptr, 0, Dd, nullptr, (int)Mr, Dd, Dd);
    gemm_bf16_wmma<0><<<dim3(cdiv(Dd, 128), cdiv(Mr, 128), 1), blk, 0, stream>>>(
        xb, 0, Dd, SDm, 0, Dd, Y2, nullptr, 0, Dd, nullptr, (int)Mr, Dd, Dd);
    // --- mix over B (32-point DFT real part) + residual ---
    mix_residual<<<dim3(cdiv(Dd, 256), Ss), blk, 0, stream>>>(Tb, Y1, Y2, x, Dd);
    // --- RMSNorm 1 -> X1 (fp32 residual) + x1b (bf16) ---
    rmsnorm_kernel<<<(unsigned)Mr, blk, 0, stream>>>(Tb, nullptr, n1s, X1, x1b, Dd);

    // --- FFN: Hb = gelu(x1b @ w1.T + b1); X2 = Hb @ w2.T + b2 ---
    gemm_bf16_wmma<2><<<dim3(cdiv(Hh, 128), cdiv(Mr, 128), 1), blk, 0, stream>>>(
        x1b, 0, Dd, w1b, 0, Dd, nullptr, Hb, 0, Hh, b1, (int)Mr, Hh, Dd);
    gemm_bf16_wmma<0><<<dim3(cdiv(Dd, 128), cdiv(Mr, 128), 1), blk, 0, stream>>>(
        Hb, 0, Hh, w2b, 0, Hh, X2, nullptr, 0, Dd, b2, (int)Mr, Dd, Hh);
    // --- RMSNorm 2 -> emotions (bf16) + transposed copy ---
    rmsnorm_kernel<<<(unsigned)Mr, blk, 0, stream>>>(X2, X1, n2s, nullptr, eb, Dd);
    make_ebT<<<4096, blk, 0, stream>>>(ebT, eb);

    // --- q = emotions @ att_w.T + att_b ---
    gemm_bf16_wmma<1><<<dim3(cdiv(Dd, 128), cdiv(Mr, 128), 1), blk, 0, stream>>>(
        eb, 0, Dd, awb, 0, Dd, nullptr, qb, 0, Dd, att_b, (int)Mr, Dd, Dd);

    // --- scores[b,t,s] = sum_d q[t,b,d]*E[s,b,d]  (batched over b) ---
    gemm_bf16_wmma<0><<<dim3(cdiv(Ss, 128), cdiv(Ss, 128), Bb), blk, 0, stream>>>(
        qb, Dd, (long)Bb * Dd,          // A: base b*D, row stride B*D
        eb, Dd, (long)Bb * Dd,          // B: elem(d,s) = eb[b*D + s*B*D + d]
        scor, nullptr, (long)Ss * Ss, Ss, nullptr, Ss, Ss, Dd);

    // --- softmax(tanh) + mask renorm -> alpha (d_out) + ab (bf16) ---
    softmax_tanh_mask<<<dim3(Ss, Bb), blk, 0, stream>>>(scor, mask, alpha, ab, Bb, Ss, Ss);

    // --- att_emotions[t,b,d] = sum_s a[b,t,s]*E[s,b,d]  (batched over b) ---
    gemm_bf16_wmma<1><<<dim3(cdiv(Dd, 128), cdiv(Ss, 128), Bb), blk, 0, stream>>>(
        ab, Ss, (long)Bb * Ss,          // A: a[t][b][s], base b*S, row stride B*S
        ebT, (long)Dd * Ss, Ss,         // B: elem(s,d) = ebT[b*D*S + d*S + s]
        nullptr, attb, Dd, (long)Bb * Dd, nullptr, Ss, Dd, Ss);

    // --- hidden = relu(att @ lin_w.T + lin_b) ---
    gemm_bf16_wmma<3><<<dim3(cdiv(Dd, 128), cdiv(Mr, 128), 1), blk, 0, stream>>>(
        attb, 0, Dd, lwb, 0, Dd, nullptr, hidb, 0, Dd, lin_b, (int)Mr, Dd, Dd);

    // --- log_prob = hidden @ fc_w.T + fc_b ---
    fc_kernel<<<(unsigned)Mr, blk, 0, stream>>>(hidb, fc_w, fc_b, log_prob, Dd);
}
